// DT_Critic_70798240907792
// MI455X (gfx1250) — matrix operand order repaired and checked by reading, not compiled, not run on hardware
//
#include <hip/hip_runtime.h>
#include <math.h>

typedef _Float16 f16;
typedef __attribute__((ext_vector_type(16))) _Float16 v16h;
typedef __attribute__((ext_vector_type(8)))  _Float16 h8;
typedef __attribute__((ext_vector_type(8)))  float    v8f;
typedef int vs4i __attribute__((vector_size(16)));

#define Bz   16
#define Tt   256
#define S3   768
#define Hh   512
#define NHh  8
#define HDd  64
#define Ll   6
#define ROWS (Bz * S3)      // 12288
#define EROWS (Bz * Tt)     // 4096
#define HH   (Hh * Hh)      // 262144

// ---------- CDNA5 async global->LDS path (guarded; falls back to sync copy) ----------
#if __has_builtin(__builtin_amdgcn_global_load_async_to_lds_b128) && \
    __has_builtin(__builtin_amdgcn_s_wait_asynccnt)
#define HAS_ASYNC_LDS 1
#else
#define HAS_ASYNC_LDS 0
#endif

#define AS_GLOBAL __attribute__((address_space(1)))
#define AS_LDS    __attribute__((address_space(3)))

__device__ __forceinline__ void cp16_g2l(const f16* gsrc, f16* ldst) {
#if HAS_ASYNC_LDS
    __builtin_amdgcn_global_load_async_to_lds_b128(
        (AS_GLOBAL vs4i*)gsrc, (AS_LDS vs4i*)ldst, 0, 0);
#else
    *(uint4*)ldst = *(const uint4*)gsrc;
#endif
}
template <int N>
__device__ __forceinline__ void wait_async_le() {
#if HAS_ASYNC_LDS
    __builtin_amdgcn_s_wait_asynccnt(N);
#endif
}

// ---------------- WMMA fragment helpers (wave32, 16x16x32 f16) ----------------
// A 16x32 (MxK): lane L holds row M=L&15; halves j -> K=(L>>4)*16+j
__device__ __forceinline__ v16h frag_a(const f16* __restrict__ base, int stride, int lane) {
    const f16* p = base + (size_t)(lane & 15) * stride + (lane >> 4) * 16;
    union { v16h v; h8 h[2]; } u;
    u.h[0] = *(const h8*)(p);
    u.h[1] = *(const h8*)(p + 8);
    return u.v;
}
// B 32x16 (KxN) sourced from N-major storage Bs[n][k]: lane L holds col N=L&15;
// halves j -> K = (L>>4)*16 + j   (mirrors the documented sparse-B layout)
__device__ __forceinline__ v16h frag_b(const f16* __restrict__ base, int stride, int lane) {
    const f16* p = base + (size_t)(lane & 15) * stride + (lane >> 4) * 16;
    union { v16h v; h8 h[2]; } u;
    u.h[0] = *(const h8*)(p);
    u.h[1] = *(const h8*)(p + 8);
    return u.v;
}
// B 32x16 fragment from row-major K x N storage (strided gather), for P.V with V rows = keys
__device__ __forceinline__ v16h frag_b_rowmajor(const f16* __restrict__ base, int rowstride, int lane) {
    const f16* p = base + (size_t)((lane >> 4) * 16) * rowstride + (lane & 15);
    v16h v;
#pragma unroll
    for (int j = 0; j < 16; ++j) v[j] = p[(size_t)j * rowstride];
    return v;
}

// ---------------- weight convert ----------------
__global__ void f32_to_f16_kn(const float* __restrict__ in, f16* __restrict__ out, int n) {
    int i = blockIdx.x * 256 + threadIdx.x;
    if (i < n) out[i] = (f16)in[i];
}

// ---------------- fused token-embed + time concat ----------------
__global__ __launch_bounds__(128)
void build_cat_kn(const float* __restrict__ src, int Kdim,
                  const float* __restrict__ W, const float* __restrict__ bias,
                  const float* __restrict__ time_table, const int* __restrict__ timesteps,
                  f16* __restrict__ cat) {
    int row = blockIdx.x;                 // b*T + t
    __shared__ float sv[32];
    int tid = threadIdx.x;
    if (tid < Kdim) sv[tid] = src[(size_t)row * Kdim + tid];
    __syncthreads();
    const float* te = time_table + (size_t)timesteps[row] * Hh;
    f16* crow = cat + (size_t)row * (2 * Hh);
    for (int c = tid; c < Hh; c += 128) {
        float acc = bias[c];
        for (int k = 0; k < Kdim; ++k) acc += sv[k] * W[(size_t)k * Hh + c];
        crow[c] = (f16)acc;
        crow[Hh + c] = (f16)te[c];
    }
}

// ---------------- tiled WMMA GEMM: C(MxN) = A(MxK,f16) * B(KxN,f16) + epilogue ----------------
#define BM 128
#define BN 128
#define BK 32
#define LDA 40   // halves; 80B rows keep 16B alignment, stride-pad vs banks

enum { EP_F32_BIAS = 0, EP_F16_BIAS = 1, EP_F32_BIAS_RES = 2, EP_F16_BIAS_GELU = 3 };

__global__ __launch_bounds__(128)
void wmma_gemm_kn(const f16* __restrict__ A, const f16* __restrict__ Bw,
                  const float* __restrict__ bias, const float* __restrict__ res,
                  void* __restrict__ Cout, int M, int N, int K, int mode) {
    __shared__ __align__(16) f16 Ash[2][BM * LDA];
    __shared__ __align__(16) f16 Bsh[2][BN * LDA];
    const int tid  = threadIdx.x;
    const int lane = tid & 31, wave = tid >> 5;
    const int wr = wave >> 1, wc = wave & 1;
    const int m0 = blockIdx.y * BM, n0 = blockIdx.x * BN;

    // stage one K-tile (A via async global->LDS DMA when available, B transposed N-major)
    auto load_tile = [&](int k0, int buf) {
        const f16* srcA = A + (size_t)(m0 + tid) * K + k0;
        f16* dA = &Ash[buf][tid * LDA];
        cp16_g2l(srcA,      dA);
        cp16_g2l(srcA + 8,  dA + 8);
        cp16_g2l(srcA + 16, dA + 16);
        cp16_g2l(srcA + 24, dA + 24);
        if (k0 + BK < K) __builtin_prefetch(srcA + BK, 0, 1);   // global_prefetch_b8
        const f16* srcB = Bw + (size_t)k0 * N + (n0 + tid);
        f16* dB = &Bsh[buf][tid * LDA];
#pragma unroll
        for (int kk = 0; kk < BK; ++kk)
            dB[kk] = srcB[(size_t)kk * N];
    };

    v8f acc[4][4] = {};
    const int nk = K / BK;
    load_tile(0, 0);
    for (int kt = 0; kt < nk; ++kt) {
        const int buf = kt & 1;
        if (kt + 1 < nk) {
            load_tile((kt + 1) * BK, buf ^ 1);
            // tile kt's 4 async fills done once <=4 remain (tile kt+1 in flight)
            wait_async_le<4>();
        } else {
            wait_async_le<0>();
        }
        __syncthreads();

        v16h aF[4], bF[4];
#pragma unroll
        for (int mi = 0; mi < 4; ++mi) aF[mi] = frag_a(&Ash[buf][(wr * 64 + mi * 16) * LDA], LDA, lane);
#pragma unroll
        for (int ni = 0; ni < 4; ++ni) bF[ni] = frag_b(&Bsh[buf][(wc * 64 + ni * 16) * LDA], LDA, lane);
#pragma unroll
        for (int mi = 0; mi < 4; ++mi)
#pragma unroll
            for (int ni = 0; ni < 4; ++ni)
                acc[mi][ni] = __builtin_amdgcn_wmma_f32_16x16x32_f16(
                    false, aF[mi], false, bF[ni], (short)0, acc[mi][ni], false, false);
        __syncthreads();   // LDS reads done before buffer reuse next-next tile
    }

    // epilogue: C layout — lane holds rows {(lane>>4)*8 + r}, col = lane&15
    const int rb = (lane >> 4) * 8, cl = lane & 15;
#pragma unroll
    for (int mi = 0; mi < 4; ++mi)
#pragma unroll
        for (int ni = 0; ni < 4; ++ni) {
            const int colg = n0 + wc * 64 + ni * 16 + cl;
            const float bs = bias[colg];
#pragma unroll
            for (int r = 0; r < 8; ++r) {
                const int rowg = m0 + wr * 64 + mi * 16 + rb + r;
                const size_t off = (size_t)rowg * N + colg;
                float v = acc[mi][ni][r] + bs;
                if      (mode == EP_F32_BIAS)     ((float*)Cout)[off] = v;
                else if (mode == EP_F16_BIAS)     ((f16*)Cout)[off]   = (f16)v;
                else if (mode == EP_F32_BIAS_RES) ((float*)Cout)[off] = v + res[off];
                else { float g = 0.5f * v * (1.0f + erff(v * 0.70710678118f));
                       ((f16*)Cout)[off] = (f16)g; }
            }
        }
}

// ---------------- interleave (re,se,ae) + LN0 -> x (fp32) ----------------
__global__ __launch_bounds__(256)
void interleave_ln0_kn(const float* __restrict__ re, const float* __restrict__ se,
                       const float* __restrict__ ae, const float* __restrict__ g,
                       const float* __restrict__ bb, float* __restrict__ x) {
    int j = blockIdx.x;                        // 0..ROWS-1
    int b = j / S3, s = j % S3, t = s / 3, kind = s % 3;
    const float* src = (kind == 0 ? re : kind == 1 ? se : ae) + (size_t)(b * Tt + t) * Hh;
    __shared__ float red[256];
    int tid = threadIdx.x;
    float v0 = src[tid], v1 = src[tid + 256];
    red[tid] = v0 + v1; __syncthreads();
    for (int st = 128; st > 0; st >>= 1) { if (tid < st) red[tid] += red[tid + st]; __syncthreads(); }
    float mean = red[0] * (1.0f / Hh); __syncthreads();
    float d0 = v0 - mean, d1 = v1 - mean;
    red[tid] = d0 * d0 + d1 * d1; __syncthreads();
    for (int st = 128; st > 0; st >>= 1) { if (tid < st) red[tid] += red[tid + st]; __syncthreads(); }
    float rstd = rsqrtf(red[0] * (1.0f / Hh) + 1e-5f);
    float* dst = x + (size_t)j * Hh;
    dst[tid]       = d0 * rstd * g[tid]       + bb[tid];
    dst[tid + 256] = d1 * rstd * g[tid + 256] + bb[tid + 256];
}

// ---------------- LN(x) -> f16 ----------------
__global__ __launch_bounds__(256)
void ln_f16_kn(const float* __restrict__ x, const float* __restrict__ g,
               const float* __restrict__ bb, f16* __restrict__ out) {
    int j = blockIdx.x;
    const float* src = x + (size_t)j * Hh;
    __shared__ float red[256];
    int tid = threadIdx.x;
    float v0 = src[tid], v1 = src[tid + 256];
    red[tid] = v0 + v1; __syncthreads();
    for (int st = 128; st > 0; st >>= 1) { if (tid < st) red[tid] += red[tid + st]; __syncthreads(); }
    float mean = red[0] * (1.0f / Hh); __syncthreads();
    float d0 = v0 - mean, d1 = v1 - mean;
    red[tid] = d0 * d0 + d1 * d1; __syncthreads();
    for (int st = 128; st > 0; st >>= 1) { if (tid < st) red[tid] += red[tid + st]; __syncthreads(); }
    float rstd = rsqrtf(red[0] * (1.0f / Hh) + 1e-5f);
    f16* dst = out + (size_t)j * Hh;
    dst[tid]       = (f16)(d0 * rstd * g[tid]       + bb[tid]);
    dst[tid + 256] = (f16)(d1 * rstd * g[tid + 256] + bb[tid + 256]);
}

// ---------------- flash attention: 1 wave per (b, head, 16-query tile) ----------------
__global__ __launch_bounds__(32)
void attn_kn(const f16* __restrict__ Q, const f16* __restrict__ K,
             const f16* __restrict__ V, const int* __restrict__ amask,
             f16* __restrict__ Y) {
    __shared__ __align__(16) f16 Psh[16 * 32];
    const int lane = threadIdx.x;
    const int q0 = blockIdx.x * 16, h = blockIdx.y, b = blockIdx.z;
    const f16* Qb = Q + (size_t)b * S3 * Hh + h * HDd;
    const f16* Kb = K + (size_t)b * S3 * Hh + h * HDd;
    const f16* Vb = V + (size_t)b * S3 * Hh + h * HDd;
    const v16h aQ0 = frag_a(Qb + (size_t)q0 * Hh,      Hh, lane);
    const v16h aQ1 = frag_a(Qb + (size_t)q0 * Hh + 32, Hh, lane);
    float mrow[8], lrow[8];
#pragma unroll
    for (int r = 0; r < 8; ++r) { mrow[r] = -1e30f; lrow[r] = 0.0f; }
    v8f acc[4] = {};
    const int rb = (lane >> 4) * 8, cl = lane & 15;
    const float scale = 0.125f;                         // 1/sqrt(64)

    for (int j0 = 0; j0 < q0 + 16; j0 += 32) {          // causal: keys <= q0+15
        v8f s[2];
#pragma unroll
        for (int c = 0; c < 2; ++c) {
            const int jb = j0 + c * 16;
            v16h bK0 = frag_b(Kb + (size_t)jb * Hh,      Hh, lane);
            v16h bK1 = frag_b(Kb + (size_t)jb * Hh + 32, Hh, lane);
            v8f sc = {};
            sc = __builtin_amdgcn_wmma_f32_16x16x32_f16(false, aQ0, false, bK0, (short)0, sc, false, false);
            sc = __builtin_amdgcn_wmma_f32_16x16x32_f16(false, aQ1, false, bK1, (short)0, sc, false, false);
            const int kj = jb + cl;
            const float madd = (1.0f - (float)amask[b * Tt + kj / 3]) * -10000.0f;
#pragma unroll
            for (int r = 0; r < 8; ++r) {
                const int qi = q0 + rb + r;
                float sv = (kj <= qi) ? (float)sc[r] * scale : -10000.0f;
                s[c][r] = sv + madd;
            }
        }
        float fac[8];
#pragma unroll
        for (int r = 0; r < 8; ++r) {
            float mx = fmaxf(s[0][r], s[1][r]);
            for (int msk = 8; msk >= 1; msk >>= 1) mx = fmaxf(mx, __shfl_xor(mx, msk, 16));
            float mnew = fmaxf(mrow[r], mx);
            fac[r] = __expf(mrow[r] - mnew);
            float p0 = __expf(s[0][r] - mnew);
            float p1 = __expf(s[1][r] - mnew);
            float ps = p0 + p1;
            for (int msk = 8; msk >= 1; msk >>= 1) ps += __shfl_xor(ps, msk, 16);
            lrow[r] = lrow[r] * fac[r] + ps;
            mrow[r] = mnew;
            Psh[(rb + r) * 32 + cl]      = (f16)p0;
            Psh[(rb + r) * 32 + 16 + cl] = (f16)p1;
        }
        __syncthreads();
        const v16h aP = frag_a(Psh, 32, lane);
#pragma unroll
        for (int e = 0; e < 4; ++e) {
#pragma unroll
            for (int r = 0; r < 8; ++r) acc[e][r] *= fac[r];
            v16h bV = frag_b_rowmajor(Vb + (size_t)j0 * Hh + e * 16, Hh, lane);
            acc[e] = __builtin_amdgcn_wmma_f32_16x16x32_f16(false, aP, false, bV, (short)0, acc[e], false, false);
        }
        __syncthreads();
    }
#pragma unroll
    for (int e = 0; e < 4; ++e)
#pragma unroll
        for (int r = 0; r < 8; ++r) {
            const int row = q0 + rb + r;
            Y[((size_t)b * S3 + row) * Hh + h * HDd + e * 16 + cl] = (f16)(acc[e][r] / lrow[r]);
        }
}

// ---------------- final scatter: out[b][kind][t][h] = x[b][3t+kind][h] ----------------
__global__ void final_out_kn(const float* __restrict__ x, float* __restrict__ out) {
    int idx = blockIdx.x * 256 + threadIdx.x;      // ROWS*H total
    int hcol = idx & (Hh - 1);
    int rest = idx >> 9;                           // b*S3 + s
    int s = rest % S3, b = rest / S3;
    int t = s / 3, kind = s % 3;
    out[(((size_t)b * 3 + kind) * Tt + t) * Hh + hcol] = x[idx];
}

// =====================================================================
extern "C" void kernel_launch(void* const* d_in, const int* in_sizes, int n_in,
                              void* d_out, int out_size, void* d_ws, size_t ws_size,
                              hipStream_t stream) {
    (void)in_sizes; (void)n_in; (void)out_size; (void)ws_size;
    const float* states   = (const float*)d_in[0];
    const float* actions  = (const float*)d_in[1];
    const float* rtg      = (const float*)d_in[3];
    const int*   tsteps   = (const int*)d_in[6];
    const int*   amask    = (const int*)d_in[7];
    const float* W_es = (const float*)d_in[8],  *b_es = (const float*)d_in[9];
    const float* W_ea = (const float*)d_in[10], *b_ea = (const float*)d_in[11];
    const float* W_er = (const float*)d_in[12], *b_er = (const float*)d_in[13];
    const float* time_table = (const float*)d_in[14];
    const float* W_ts = (const float*)d_in[15], *b_ts = (const float*)d_in[16];
    const float* W_ta = (const float*)d_in[17], *b_ta = (const float*)d_in[18];
    const float* W_tr = (const float*)d_in[19], *b_tr = (const float*)d_in[20];
    const float* ln0_g = (const float*)d_in[21], *ln0_b = (const float*)d_in[22];
    const float* ln1_g = (const float*)d_in[23], *ln1_b = (const float*)d_in[24];
    const float* ln2_g = (const float*)d_in[25], *ln2_b = (const float*)d_in[26];
    const float* Wq = (const float*)d_in[27], *bq = (const float*)d_in[28];
    const float* Wk = (const float*)d_in[29], *bk = (const float*)d_in[30];
    const float* Wv = (const float*)d_in[31], *bv = (const float*)d_in[32];
    const float* Wo = (const float*)d_in[33], *bo = (const float*)d_in[34];
    const float* Wm1 = (const float*)d_in[35], *bm1 = (const float*)d_in[36];
    const float* Wm2 = (const float*)d_in[37], *bm2 = (const float*)d_in[38];

    char* ws = (char*)d_ws;
    // f16 weight area
    f16* wts16 = (f16*)(ws + 0);
    f16* wta16 = (f16*)(ws + 1048576);
    f16* wtr16 = (f16*)(ws + 2097152);
    f16* wq16  = (f16*)(ws + 3145728);
    f16* wk16  = (f16*)(ws + 6291456);
    f16* wv16  = (f16*)(ws + 9437184);
    f16* wo16  = (f16*)(ws + 12582912);
    f16* wm116 = (f16*)(ws + 15728640);
    f16* wm216 = (f16*)(ws + 18874368);
    // activations
    float* x  = (float*)(ws + 22020096);          // ROWS*H f32
    f16*   xn = (f16*)  (ws + 47185920);          // ROWS*H f16
    f16*   q  = (f16*)  (ws + 59768832);
    f16*   k  = (f16*)  (ws + 72351744);
    f16*   v  = (f16*)  (ws + 84934656);
    f16*   y  = (f16*)  (ws + 97517568);
    f16*   tb = (f16*)  (ws + 110100480);
    // aliases (embedding phase only)
    f16* cat_s = q;  f16* cat_a = k;  f16* cat_r = v;
    float* re = (float*)xn;  float* se = (float*)y;  float* ae = (float*)tb;

    // 1) weights -> f16
    auto cvt = [&](const float* src, f16* dst, int n) {
        f32_to_f16_kn<<<(n + 255) / 256, 256, 0, stream>>>(src, dst, n);
    };
    cvt(W_ts, wts16, 2 * HH); cvt(W_ta, wta16, 2 * HH); cvt(W_tr, wtr16, 2 * HH);
    cvt(Wq, wq16, Ll * HH); cvt(Wk, wk16, Ll * HH); cvt(Wv, wv16, Ll * HH);
    cvt(Wo, wo16, Ll * HH); cvt(Wm1, wm116, Ll * HH); cvt(Wm2, wm216, Ll * HH);

    // 2) token embeddings + time concat
    build_cat_kn<<<EROWS, 128, 0, stream>>>(states,  17, W_es, b_es, time_table, tsteps, cat_s);
    build_cat_kn<<<EROWS, 128, 0, stream>>>(actions,  6, W_ea, b_ea, time_table, tsteps, cat_a);
    build_cat_kn<<<EROWS, 128, 0, stream>>>(rtg,      1, W_er, b_er, time_table, tsteps, cat_r);

    // 3) time-fusion GEMMs (M=4096, K=1024, N=512)
    dim3 ge(Hh / BN, EROWS / BM);
    wmma_gemm_kn<<<ge, 128, 0, stream>>>(cat_s, wts16, b_ts, nullptr, se, EROWS, Hh, 2 * Hh, EP_F32_BIAS);
    wmma_gemm_kn<<<ge, 128, 0, stream>>>(cat_a, wta16, b_ta, nullptr, ae, EROWS, Hh, 2 * Hh, EP_F32_BIAS);
    wmma_gemm_kn<<<ge, 128, 0, stream>>>(cat_r, wtr16, b_tr, nullptr, re, EROWS, Hh, 2 * Hh, EP_F32_BIAS);

    // 4) interleave + ln0
    interleave_ln0_kn<<<ROWS, 256, 0, stream>>>(re, se, ae, ln0_g, ln0_b, x);

    // 5) transformer layers
    dim3 gp(Hh / BN, ROWS / BM);
    dim3 ga(S3 / 16, NHh, Bz);
    for (int l = 0; l < Ll; ++l) {
        ln_f16_kn<<<ROWS, 256, 0, stream>>>(x, ln1_g + l * Hh, ln1_b + l * Hh, xn);
        wmma_gemm_kn<<<gp, 128, 0, stream>>>(xn, wq16 + (size_t)l * HH, bq + l * Hh, nullptr, q, ROWS, Hh, Hh, EP_F16_BIAS);
        wmma_gemm_kn<<<gp, 128, 0, stream>>>(xn, wk16 + (size_t)l * HH, bk + l * Hh, nullptr, k, ROWS, Hh, Hh, EP_F16_BIAS);
        wmma_gemm_kn<<<gp, 128, 0, stream>>>(xn, wv16 + (size_t)l * HH, bv + l * Hh, nullptr, v, ROWS, Hh, Hh, EP_F16_BIAS);
        attn_kn<<<ga, 32, 0, stream>>>(q, k, v, amask, y);
        wmma_gemm_kn<<<gp, 128, 0, stream>>>(y, wo16 + (size_t)l * HH, bo + l * Hh, x, x, ROWS, Hh, Hh, EP_F32_BIAS_RES);
        ln_f16_kn<<<ROWS, 256, 0, stream>>>(x, ln2_g + l * Hh, ln2_b + l * Hh, xn);
        wmma_gemm_kn<<<gp, 128, 0, stream>>>(xn, wm116 + (size_t)l * HH, bm1 + l * Hh, nullptr, tb, ROWS, Hh, Hh, EP_F16_BIAS_GELU);
        wmma_gemm_kn<<<gp, 128, 0, stream>>>(tb, wm216 + (size_t)l * HH, bm2 + l * Hh, x, x, ROWS, Hh, Hh, EP_F32_BIAS_RES);
    }

    // 6) output scatter (B,3,T,H)
    final_out_kn<<<(ROWS * Hh) / 256, 256, 0, stream>>>(x, (float*)d_out);
}
